// GTrXL_55336358642569
// MI455X (gfx1250) — matrix operand order repaired and checked by reading, not compiled, not run on hardware
//
#include <hip/hip_runtime.h>
#include <hip/hip_bf16.h>

// ---------------------------------------------------------------------------
// GTrXL layer for gfx1250 (MI455X).  All dense GEMMs run on the matrix pipe
// via v_wmma_f32_16x16x32_bf16 (bf16 inputs, f32 accumulate), with
// double-buffered LDS staging so global prefetch overlaps the WMMA pipe.
// ---------------------------------------------------------------------------

typedef __attribute__((ext_vector_type(16))) __bf16 v16bf;
typedef __attribute__((ext_vector_type(8)))  float  v8f;

#define CUR_  512
#define PREV_ 512
#define FULL_ 1024
#define BS_   8
#define HN_   16
#define HD_   64
#define D_    1024
#define HID_  4096
#define SCALE_ 0.125f   // 1/sqrt(64)

__device__ __forceinline__ unsigned short f2bf(float f) {
  union { float f; unsigned int u; } c; c.f = f;
  unsigned int r = c.u + 0x7FFFu + ((c.u >> 16) & 1u);   // round-to-nearest-even
  return (unsigned short)(r >> 16);
}

// ---- stage a 64x32 f32 tile (rows x K) as bf16 into LDS [r][k] ------------
__device__ __forceinline__ void stage_rows_k(
    unsigned short* __restrict__ dst, const float* __restrict__ src,
    const float* __restrict__ addp, int r0, int k0, int ld, int tid)
{
#pragma unroll
  for (int i = 0; i < 4; ++i) {
    int id = tid + i * 128;
    int r  = id >> 3;          // 0..63
    int c4 = id & 7;           // float4 index along k
    const float* p = src + (long long)(r0 + r) * ld + (k0 + c4 * 4);
    float4 v = *(const float4*)p;
    if (addp) {
      const float* q = addp + k0 + c4 * 4;
      v.x += q[0]; v.y += q[1]; v.z += q[2]; v.w += q[3];
    }
    uint2 pk;
    pk.x = (unsigned)f2bf(v.x) | ((unsigned)f2bf(v.y) << 16);
    pk.y = (unsigned)f2bf(v.z) | ((unsigned)f2bf(v.w) << 16);
    *(uint2*)&dst[r * 32 + c4 * 4] = pk;
  }
}

// ---- stage a 32(K) x 64(N) f32 tile transposed into LDS [n][k] ------------
__device__ __forceinline__ void stage_k_rows(
    unsigned short* __restrict__ dst, const float* __restrict__ src,
    int n0, int k0, int ld, int tid)
{
#pragma unroll
  for (int i = 0; i < 4; ++i) {
    int id = tid + i * 128;
    int kk = id >> 4;          // 0..31
    int c4 = id & 15;          // float4 index along n
    const float* p = src + (long long)(k0 + kk) * ld + (n0 + c4 * 4);
    float4 v = *(const float4*)p;
    int nb = c4 * 4;
    dst[(nb + 0) * 32 + kk] = f2bf(v.x);
    dst[(nb + 1) * 32 + kk] = f2bf(v.y);
    dst[(nb + 2) * 32 + kk] = f2bf(v.z);
    dst[(nb + 3) * 32 + kk] = f2bf(v.w);
  }
}

// ---------------------------------------------------------------------------
// Generic tiled GEMM:  C[M,N] = act( A_eff[M,K] @ B[K,N] + bias )
//   A_eff[m,k] = A[m,k] + (aadd ? aadd[k] : 0)
//   B_NXK:  B stored as N rows of K contiguous values (A @ B^T style)
//   batch z (gridDim.z):  off = (z/zdiv)*s1 + (z%zdiv)*s2 for A/B/C, aadd: sAdd2
// Block: 128 threads = 4 waves; tile 64x64, K-step 32, double-buffered LDS.
// Wave w computes rows [16w,16w+16) x 64 cols -> 4 accumulators.
// ---------------------------------------------------------------------------
template <bool B_NXK, bool RELU>
__global__ __launch_bounds__(128) void gemm_bf16_wmma(
    const float* __restrict__ A, const float* __restrict__ B,
    const float* __restrict__ bias, const float* __restrict__ aadd,
    float* __restrict__ C,
    int M, int N, int K, int lda, int ldb, int ldc,
    long long sA1, long long sA2, long long sB1, long long sB2,
    long long sC1, long long sC2, long long sAdd2, int zdiv)
{
  __shared__ __align__(16) unsigned short As[2][64 * 32];
  __shared__ __align__(16) unsigned short Bs[2][64 * 32];

  const int tid  = threadIdx.x;
  const int lane = tid & 31;
  const int wv   = tid >> 5;

  const long long z  = blockIdx.z;
  const long long zq = z / zdiv;
  const long long zr = z % zdiv;
  const float* Ab = A + zq * sA1 + zr * sA2;
  const float* Bb = B + zq * sB1 + zr * sB2;
  float*       Cb = C + zq * sC1 + zr * sC2;
  const float* addp = aadd ? (aadd + zr * sAdd2) : nullptr;

  const int n0 = blockIdx.x * 64;
  const int m0 = blockIdx.y * 64;

  v8f acc[4];
#pragma unroll
  for (int t = 0; t < 4; ++t)
#pragma unroll
    for (int i = 0; i < 8; ++i) acc[t][i] = 0.f;

  // prologue: fill buffer 0
  stage_rows_k(As[0], Ab, addp, m0, 0, lda, tid);
  if (B_NXK) stage_rows_k(Bs[0], Bb, nullptr, n0, 0, ldb, tid);
  else       stage_k_rows(Bs[0], Bb, n0, 0, ldb, tid);
  __syncthreads();

  const int nk = K >> 5;
  for (int ki = 0; ki < nk; ++ki) {
    const int cur = ki & 1;
    const unsigned short* Ac = As[cur];
    const unsigned short* Bc = Bs[cur];

    // gather this slice's fragments first (ds loads issue early)
    v16bf afrag;
    {
      int m  = wv * 16 + (lane & 15);
      int kb = (lane < 16) ? 0 : 8;
      uint4* au = (uint4*)&afrag;
      au[0] = *(const uint4*)&Ac[m * 32 + kb];        // K = kb..kb+7
      au[1] = *(const uint4*)&Ac[m * 32 + kb + 16];   // K = kb+16..kb+23
    }
    v16bf bfrag[4];
#pragma unroll
    for (int t = 0; t < 4; ++t) {
      int n  = t * 16 + (lane & 15);
      int kb = (lane < 16) ? 0 : 8;
      uint4* bu = (uint4*)&bfrag[t];
      bu[0] = *(const uint4*)&Bc[n * 32 + kb];        // K = kb..kb+7
      bu[1] = *(const uint4*)&Bc[n * 32 + kb + 16];   // K = kb+16..kb+23
    }

    // prefetch next slice into the other buffer (overlaps with WMMAs)
    if (ki + 1 < nk) {
      const int k1 = (ki + 1) << 5;
      stage_rows_k(As[cur ^ 1], Ab, addp, m0, k1, lda, tid);
      if (B_NXK) stage_rows_k(Bs[cur ^ 1], Bb, nullptr, n0, k1, ldb, tid);
      else       stage_k_rows(Bs[cur ^ 1], Bb, n0, k1, ldb, tid);
    }

    // 4 back-to-back WMMAs on the matrix pipe
#pragma unroll
    for (int t = 0; t < 4; ++t)
      acc[t] = __builtin_amdgcn_wmma_f32_16x16x32_bf16(
          false, afrag, false, bfrag[t], (short)0, acc[t], false, false);

    __syncthreads();
  }

  // ---- epilogue: C/D layout -> VGPR i: lanes0-15 M=i, lanes16-31 M=i+8 ----
  const int nl = lane & 15;
  const int mh = (lane < 16) ? 0 : 8;
#pragma unroll
  for (int t = 0; t < 4; ++t) {
    int col = n0 + t * 16 + nl;
    float bv = bias ? bias[col] : 0.f;
#pragma unroll
    for (int i = 0; i < 8; ++i) {
      int row = m0 + wv * 16 + mh + i;
      float v = acc[t][i] + bv;
      if (RELU) v = fmaxf(v, 0.f);
      Cb[(long long)row * ldc + col] = v;
    }
  }
}

// ---------------------------------------------------------------------------
// LayerNorm over D=1024 with fused concat: rows < splitRows read inA,
// rows >= splitRows read inB at (row - splitRows).
// ---------------------------------------------------------------------------
__global__ __launch_bounds__(256) void ln_kernel(
    const float* __restrict__ inA, const float* __restrict__ inB, int splitRows,
    const float* __restrict__ g, const float* __restrict__ b,
    float* __restrict__ out)
{
  const int row = blockIdx.x;
  const int tid = threadIdx.x;
  const float* src = (row < splitRows) ? (inA + (long long)row * D_)
                                       : (inB + (long long)(row - splitRows) * D_);
  float4 v = ((const float4*)src)[tid];
  float s  = v.x + v.y + v.z + v.w;
  float s2 = v.x * v.x + v.y * v.y + v.z * v.z + v.w * v.w;
  __shared__ float rs[256], rs2[256];
  rs[tid] = s; rs2[tid] = s2;
  __syncthreads();
  for (int o = 128; o > 0; o >>= 1) {
    if (tid < o) { rs[tid] += rs[tid + o]; rs2[tid] += rs2[tid + o]; }
    __syncthreads();
  }
  float mean = rs[0] * (1.f / D_);
  float var  = rs2[0] * (1.f / D_) - mean * mean;
  float rstd = rsqrtf(var + 1e-5f);
  int c = tid * 4;
  float4 o4;
  o4.x = (v.x - mean) * rstd * g[c + 0] + b[c + 0];
  o4.y = (v.y - mean) * rstd * g[c + 1] + b[c + 1];
  o4.z = (v.z - mean) * rstd * g[c + 2] + b[c + 2];
  o4.w = (v.w - mean) * rstd * g[c + 3] + b[c + 3];
  ((float4*)(out + (long long)row * D_))[tid] = o4;
}

// ---------------------------------------------------------------------------
// logits = (content[s,f] + pos[s, f+CUR-1-s]) * SCALE for f <= s+PREV,
// -inf otherwise (mask + rel_shift folded in); softmax in-place into content.
// grid = (CUR, BS*HN), block = 256, FULL = 1024 -> 4 elems/thread.
// ---------------------------------------------------------------------------
__global__ __launch_bounds__(256) void attn_softmax(
    float* __restrict__ content, const float* __restrict__ posraw)
{
  const int s = blockIdx.x;
  const long long rz = (long long)blockIdx.y * CUR_ + s;
  float*       crow = content + rz * FULL_;
  const float* prow = posraw  + rz * FULL_;
  const int tid   = threadIdx.x;
  const int limit = s + PREV_;
  const int shift = CUR_ - 1 - s;

  float vals[4];
  float mx = -3.4e38f;
#pragma unroll
  for (int i = 0; i < 4; ++i) {
    int f = tid + i * 256;
    float v = -3.4e38f;
    if (f <= limit) v = (crow[f] + prow[f + shift]) * SCALE_;
    vals[i] = v;
    mx = fmaxf(mx, v);
  }
  __shared__ float red[256];
  red[tid] = mx; __syncthreads();
  for (int o = 128; o > 0; o >>= 1) {
    if (tid < o) red[tid] = fmaxf(red[tid], red[tid + o]);
    __syncthreads();
  }
  mx = red[0];
  __syncthreads();
  float sum = 0.f;
#pragma unroll
  for (int i = 0; i < 4; ++i) {
    float e = (vals[i] > -3.0e38f) ? __expf(vals[i] - mx) : 0.f;
    vals[i] = e; sum += e;
  }
  red[tid] = sum; __syncthreads();
  for (int o = 128; o > 0; o >>= 1) {
    if (tid < o) red[tid] += red[tid + o];
    __syncthreads();
  }
  float inv = 1.f / red[0];
#pragma unroll
  for (int i = 0; i < 4; ++i) crow[tid + i * 256] = vals[i] * inv;
}

// ---------------------------------------------------------------------------
// GRU gate combine: r=sig(t0+t1); z=sig(t2+t3-bg); h=tanh(t4+r*t5);
// out = (1-z)*x + z*h
// ---------------------------------------------------------------------------
__global__ __launch_bounds__(256) void gru_gate(
    const float* __restrict__ x,
    const float* __restrict__ t0, const float* __restrict__ t1,
    const float* __restrict__ t2, const float* __restrict__ t3,
    const float* __restrict__ t4, const float* __restrict__ t5,
    const float* __restrict__ bg, float* __restrict__ out, int n)
{
  int i = blockIdx.x * 256 + threadIdx.x;
  if (i >= n) return;
  int c = i & (D_ - 1);
  float r  = 1.f / (1.f + __expf(-(t0[i] + t1[i])));
  float zz = 1.f / (1.f + __expf(-(t2[i] + t3[i] - bg[c])));
  float h  = tanhf(t4[i] + r * t5[i]);
  out[i] = (1.f - zz) * x[i] + zz * h;
}

// ---------------------------------------------------------------------------
extern "C" void kernel_launch(void* const* d_in, const int* in_sizes, int n_in,
                              void* d_out, int out_size, void* d_ws, size_t ws_size,
                              hipStream_t stream) {
  (void)in_sizes; (void)n_in; (void)out_size; (void)ws_size;

  const float* inputs = (const float*)d_in[0];
  const float* pos    = (const float*)d_in[1];
  const float* memory = (const float*)d_in[2];
  const float* u      = (const float*)d_in[3];
  const float* vv     = (const float*)d_in[4];
  // d_in[5] = mask (recomputed analytically)
  const float* Wkv   = (const float*)d_in[6];
  const float* bkv   = (const float*)d_in[7];
  const float* Wq    = (const float*)d_in[8];
  const float* bq    = (const float*)d_in[9];
  const float* Wpos  = (const float*)d_in[10];
  const float* bpos  = (const float*)d_in[11];
  const float* Wproj = (const float*)d_in[12];
  const float* bproj = (const float*)d_in[13];
  const float* ln1g  = (const float*)d_in[14];
  const float* ln1b  = (const float*)d_in[15];
  const float* ln2g  = (const float*)d_in[16];
  const float* ln2b  = (const float*)d_in[17];
  const float* g1Wr  = (const float*)d_in[18];
  const float* g1Ur  = (const float*)d_in[19];
  const float* g1Wz  = (const float*)d_in[20];
  const float* g1Uz  = (const float*)d_in[21];
  const float* g1Wg  = (const float*)d_in[22];
  const float* g1Ug  = (const float*)d_in[23];
  const float* g1bg  = (const float*)d_in[24];
  const float* g2Wr  = (const float*)d_in[25];
  const float* g2Ur  = (const float*)d_in[26];
  const float* g2Wz  = (const float*)d_in[27];
  const float* g2Uz  = (const float*)d_in[28];
  const float* g2Wg  = (const float*)d_in[29];
  const float* g2Ug  = (const float*)d_in[30];
  const float* g2bg  = (const float*)d_in[31];
  const float* mW1   = (const float*)d_in[32];
  const float* mb1   = (const float*)d_in[33];
  const float* mW2   = (const float*)d_in[34];
  const float* mb2   = (const float*)d_in[35];
  float* out = (float*)d_out;
  float* ws  = (float*)d_ws;

  // workspace layout (floats) with reuse
  const long long N_X1 = 8LL * 1024 * 1024;          // 8192 x 1024
  const long long N_KV = 16LL * 1024 * 1024;         // 8192 x 2048
  const long long N_Q  = 4LL * 1024 * 1024;          // 4096 x 1024
  const long long N_R  = 1LL * 1024 * 1024;          // 1024 x 1024
  const long long N_SC = 64LL * 1024 * 1024;         // 128 x 512 x 1024
  const long long N_T  = 4LL * 1024 * 1024;

  float* x1   = ws;
  float* kv   = x1 + N_X1;
  float* q    = kv + N_KV;
  float* rprj = q  + N_Q;
  float* ctnt = rprj + N_R;          // content scores -> probs -> GRU temps t0..t5
  float* prw  = ctnt + N_SC;         // raw position scores -> x2 + h1
  float* av   = prw + N_SC;
  float* a1   = av + N_T;            // later reused for m2
  float* o1   = a1 + N_T;
  // aliases (lifetimes disjoint)
  float* t[6]; for (int i = 0; i < 6; ++i) t[i] = ctnt + (long long)i * N_T;
  float* x2 = prw;
  float* h1 = prw + N_T;
  float* m2 = a1;

  const dim3 B128(128), B256(256);
  const long long SCZ = (long long)CUR_ * FULL_;     // 524288

  // 1) x1 = LN(concat(memory, inputs))
  ln_kernel<<<dim3(FULL_ * BS_), B256, 0, stream>>>(memory, inputs, PREV_ * BS_,
                                                    ln1g, ln1b, x1);
  // 2) kv = x1 @ Wkv + bkv     (8192 x 2048 x 1024)
  gemm_bf16_wmma<false, false><<<dim3(32, 128, 1), B128, 0, stream>>>(
      x1, Wkv, bkv, nullptr, kv, 8192, 2048, 1024, 1024, 2048, 2048,
      0, 0, 0, 0, 0, 0, 0, 1);
  // 3) q = x1[-cur:] @ Wq + bq (4096 x 1024 x 1024)
  gemm_bf16_wmma<false, false><<<dim3(16, 64, 1), B128, 0, stream>>>(
      x1 + (long long)PREV_ * BS_ * D_, Wq, bq, nullptr, q,
      4096, 1024, 1024, 1024, 1024, 1024, 0, 0, 0, 0, 0, 0, 0, 1);
  // 4) rproj = pos @ Wpos + bpos (1024 x 1024 x 1024)
  gemm_bf16_wmma<false, false><<<dim3(16, 16, 1), B128, 0, stream>>>(
      pos, Wpos, bpos, nullptr, rprj, 1024, 1024, 1024, 1024, 1024, 1024,
      0, 0, 0, 0, 0, 0, 0, 1);
  // 5) content[z] = (q+u) @ key^T, z = b*16+h  (batched 128: 512x1024x64)
  gemm_bf16_wmma<true, false><<<dim3(16, 8, 128), B128, 0, stream>>>(
      q, kv, nullptr, u, ctnt, 512, 1024, 64,
      /*lda*/ BS_ * HN_ * HD_, /*ldb*/ BS_ * 2048, /*ldc*/ FULL_,
      /*sA*/ HN_ * HD_, HD_, /*sB*/ 2048, HD_, /*sC*/ 16 * SCZ, SCZ,
      /*sAdd2*/ HD_, 16);
  // 6) posraw[z] = (q+v) @ r_h^T
  gemm_bf16_wmma<true, false><<<dim3(16, 8, 128), B128, 0, stream>>>(
      q, rprj, nullptr, vv, prw, 512, 1024, 64,
      BS_ * HN_ * HD_, HN_ * HD_, FULL_,
      HN_ * HD_, HD_, 0, HD_, 16 * SCZ, SCZ, HD_, 16);
  // 7) mask + rel-shift + softmax in place (ctnt becomes probabilities)
  attn_softmax<<<dim3(CUR_, BS_ * HN_), B256, 0, stream>>>(ctnt, prw);
  // 8) av[z] = P @ value  (batched 128: 512x64x1024)
  gemm_bf16_wmma<false, false><<<dim3(1, 8, 128), B128, 0, stream>>>(
      ctnt, kv + HN_ * HD_, nullptr, nullptr, av, 512, 64, 1024,
      FULL_, BS_ * 2048, BS_ * HN_ * HD_,
      16 * SCZ, SCZ, 2048, HD_, 16 * HD_, HD_, 0, 16);
  // 9) a1 = relu(av @ Wproj + bproj)  (4096 x 1024 x 1024)
  gemm_bf16_wmma<false, true><<<dim3(16, 64, 1), B128, 0, stream>>>(
      av, Wproj, bproj, nullptr, a1, 4096, 1024, 1024, 1024, 1024, 1024,
      0, 0, 0, 0, 0, 0, 0, 1);
  // 10) GRU gate 1 matmuls
  gemm_bf16_wmma<false, false><<<dim3(16, 64, 1), B128, 0, stream>>>(
      a1, g1Wr, nullptr, nullptr, t[0], 4096, 1024, 1024, 1024, 1024, 1024, 0,0,0,0,0,0,0,1);
  gemm_bf16_wmma<false, false><<<dim3(16, 64, 1), B128, 0, stream>>>(
      inputs, g1Ur, nullptr, nullptr, t[1], 4096, 1024, 1024, 1024, 1024, 1024, 0,0,0,0,0,0,0,1);
  gemm_bf16_wmma<false, false><<<dim3(16, 64, 1), B128, 0, stream>>>(
      a1, g1Wz, nullptr, nullptr, t[2], 4096, 1024, 1024, 1024, 1024, 1024, 0,0,0,0,0,0,0,1);
  gemm_bf16_wmma<false, false><<<dim3(16, 64, 1), B128, 0, stream>>>(
      inputs, g1Uz, nullptr, nullptr, t[3], 4096, 1024, 1024, 1024, 1024, 1024, 0,0,0,0,0,0,0,1);
  gemm_bf16_wmma<false, false><<<dim3(16, 64, 1), B128, 0, stream>>>(
      a1, g1Wg, nullptr, nullptr, t[4], 4096, 1024, 1024, 1024, 1024, 1024, 0,0,0,0,0,0,0,1);
  gemm_bf16_wmma<false, false><<<dim3(16, 64, 1), B128, 0, stream>>>(
      inputs, g1Ug, nullptr, nullptr, t[5], 4096, 1024, 1024, 1024, 1024, 1024, 0,0,0,0,0,0,0,1);
  // 11) o1 = gate(inputs, a1-side temps)
  gru_gate<<<dim3(16384), B256, 0, stream>>>(inputs, t[0], t[1], t[2], t[3], t[4], t[5],
                                             g1bg, o1, 4096 * 1024);
  // 12) x2 = LN2(o1)
  ln_kernel<<<dim3(CUR_ * BS_), B256, 0, stream>>>(o1, o1, CUR_ * BS_, ln2g, ln2b, x2);
  // 13) h1 = relu(x2 @ W1 + b1)  (4096 x 4096 x 1024)
  gemm_bf16_wmma<false, true><<<dim3(64, 64, 1), B128, 0, stream>>>(
      x2, mW1, mb1, nullptr, h1, 4096, 4096, 1024, 1024, 4096, 4096,
      0, 0, 0, 0, 0, 0, 0, 1);
  // 14) m2 = relu(h1 @ W2 + b2)  (4096 x 1024 x 4096)
  gemm_bf16_wmma<false, true><<<dim3(16, 64, 1), B128, 0, stream>>>(
      h1, mW2, mb2, nullptr, m2, 4096, 1024, 4096, 4096, 1024, 1024,
      0, 0, 0, 0, 0, 0, 0, 1);
  // 15) GRU gate 2 matmuls
  gemm_bf16_wmma<false, false><<<dim3(16, 64, 1), B128, 0, stream>>>(
      m2, g2Wr, nullptr, nullptr, t[0], 4096, 1024, 1024, 1024, 1024, 1024, 0,0,0,0,0,0,0,1);
  gemm_bf16_wmma<false, false><<<dim3(16, 64, 1), B128, 0, stream>>>(
      o1, g2Ur, nullptr, nullptr, t[1], 4096, 1024, 1024, 1024, 1024, 1024, 0,0,0,0,0,0,0,1);
  gemm_bf16_wmma<false, false><<<dim3(16, 64, 1), B128, 0, stream>>>(
      m2, g2Wz, nullptr, nullptr, t[2], 4096, 1024, 1024, 1024, 1024, 1024, 0,0,0,0,0,0,0,1);
  gemm_bf16_wmma<false, false><<<dim3(16, 64, 1), B128, 0, stream>>>(
      o1, g2Uz, nullptr, nullptr, t[3], 4096, 1024, 1024, 1024, 1024, 1024, 0,0,0,0,0,0,0,1);
  gemm_bf16_wmma<false, false><<<dim3(16, 64, 1), B128, 0, stream>>>(
      m2, g2Wg, nullptr, nullptr, t[4], 4096, 1024, 1024, 1024, 1024, 1024, 0,0,0,0,0,0,0,1);
  gemm_bf16_wmma<false, false><<<dim3(16, 64, 1), B128, 0, stream>>>(
      o1, g2Ug, nullptr, nullptr, t[5], 4096, 1024, 1024, 1024, 1024, 1024, 0,0,0,0,0,0,0,1);
  // 16) o2 = gate(o1, m2-side temps) -> d_out
  gru_gate<<<dim3(16384), B256, 0, stream>>>(o1, t[0], t[1], t[2], t[3], t[4], t[5],
                                             g2bg, out, 4096 * 1024);
}
